// MultiheadSelfAttention_13932873909170
// MI455X (gfx1250) — matrix operand order repaired and checked by reading, not compiled, not run on hardware
//
#include <hip/hip_runtime.h>
#include <hip/hip_bf16.h>

// Nystromformer attention for MI455X (gfx1250, wave32, WMMA + TDM).
// Large GEMMs: v_wmma_f32_16x16x32_f16 (f16 operands, fp32 accum).
// K>=128 GEMMs stage the shared 64-row W panel into LDS via the Tensor Data
// Mover (tensor_load_to_lds, TENSORcnt) with double buffering.
// Small dense algebra (softmax, landmarks, Moore-Penrose pinv) in fp32 VALU.

typedef __attribute__((ext_vector_type(16))) _Float16 v16h;
typedef __attribute__((ext_vector_type(8)))  _Float16 v8h;
typedef __attribute__((ext_vector_type(8)))  float    v8f;
typedef __attribute__((ext_vector_type(4)))  unsigned int v4u;
typedef __attribute__((ext_vector_type(8)))  int      v8i;
typedef __attribute__((ext_vector_type(4)))  int      v4i;
typedef _Float16 half_t;

#define NEGF (-100000.0f)
#define TDM_KB 128   // K-block staged per TDM transfer (64 rows x 128 cols f16 = 16KB)

// ---------------------------------------------------------------------------
// fp32 -> fp16 conversion (grid-stride)
// ---------------------------------------------------------------------------
__global__ void cvt_kernel(const float* __restrict__ in, half_t* __restrict__ out, long n) {
  long i = (long)blockIdx.x * blockDim.x + threadIdx.x;
  long stride = (long)gridDim.x * blockDim.x;
  for (; i < n; i += stride) out[i] = (half_t)in[i];
}

// ---------------------------------------------------------------------------
// per-landmark mask statistics: denom[b,m] = sum(mask) + 1e-8, mask_lm = denom>0
// ---------------------------------------------------------------------------
__global__ void mask_stats_kernel(const float* __restrict__ mask, float* __restrict__ denom,
                                  float* __restrict__ masklm, int B, int L, int nl) {
  int idx = blockIdx.x * blockDim.x + threadIdx.x;
  if (idx >= B * nl) return;
  int b = idx / nl, m = idx % nl;
  int seg = L / nl;
  const float* p = mask + (long)b * L + (long)m * seg;
  float s = 0.f;
  for (int i = 0; i < seg; ++i) s += p[i];
  float d = s + 1e-8f;
  denom[idx]  = d;
  masklm[idx] = (d > 0.f) ? 1.f : 0.f;
}

// ---------------------------------------------------------------------------
// WMMA fragment helpers (layouts per CDNA5 ISA 7.12.2)
// ---------------------------------------------------------------------------
__device__ __forceinline__ v16h load_frag_a(const half_t* p) {
  union { v16h v; v8h h[2]; } a;
  a.h[0] = *(const v8h*)(p);
  a.h[1] = *(const v8h*)(p + 16);
  return a.v;
}
__device__ __forceinline__ v16h load_frag_b(const half_t* p) {
  union { v16h v; v8h h[2]; } b;
  b.h[0] = *(const v8h*)(p);
  b.h[1] = *(const v8h*)(p + 16);
  return b.v;
}

// ---------------------------------------------------------------------------
// TDM issue: DMA a 64-row x TDM_KB-col f16 panel (row stride ldw elements)
// from global memory into LDS at byte offset lds_byte. 2D descriptor:
// group0 = {count=1 | lds_addr | global_addr(57b) | type=2}
// group1 = {data_size=2B, tensor 128x64 (== tile, no OOB), stride0=ldw}
// groups 2/3 (dims 2-4) unused; trailing v8i of the 6-arg builtin zero-filled.
// ---------------------------------------------------------------------------
__device__ __forceinline__ void tdm_issue_panel(const half_t* gptr, int ldw, unsigned lds_byte) {
  unsigned long long ga = (unsigned long long)(size_t)gptr;
  v4u g0;
  g0.x = 1u;                                               // count=1 (valid user D#)
  g0.y = lds_byte;                                         // lds_addr (bytes)
  g0.z = (unsigned)(ga & 0xFFFFFFFFu);                     // global_addr[31:0]
  g0.w = (unsigned)((ga >> 32) & 0x1FFFFFFu) | (2u << 30); // global_addr[56:32] | type=2
  v8i g1;
  g1[0] = (int)(1u << 16);                                 // data_size=1 -> 2 bytes
  g1[1] = (int)((unsigned)TDM_KB << 16);                   // tensor_dim0[15:0]
  g1[2] = (int)(64u << 16);                                // tensor_dim0[31:16]=0 | tensor_dim1[15:0]=64
  g1[3] = (int)((unsigned)TDM_KB << 16);                   // tensor_dim1[31:16]=0 | tile_dim0=TDM_KB
  g1[4] = 64;                                              // tile_dim1=64 | tile_dim2=0
  g1[5] = ldw;                                             // tensor_dim0_stride[31:0] (elements)
  g1[6] = 0;                                               // stride0[47:32] | stride1[15:0]
  g1[7] = 0;                                               // stride1[47:16]
  v4i g2 = {0, 0, 0, 0};
  v4i g3 = {0, 0, 0, 0};
  v8i g4 = {0, 0, 0, 0, 0, 0, 0, 0};
  __builtin_amdgcn_tensor_load_to_lds(g0, g1, g2, g3, g4, 0);
}

// ---------------------------------------------------------------------------
// Generic NT WMMA GEMM (direct global):  C[M,N] = A[M,K] * W[N,K]^T
// Block: 256 threads = 8 waves; block tile 64x64; wave tile 16x32.
// grid = (N/64, M/64, Z). Used for the K=64 GEMMs.
// ---------------------------------------------------------------------------
template<int C_F16, int BIAS>
__global__ __launch_bounds__(256) void gemm_nt_wmma(
    const half_t* __restrict__ A, const half_t* __restrict__ W,
    void* __restrict__ Cv, const float* __restrict__ bias,
    int K, int lda, int ldw, int ldc,
    long strideA, long strideW, int zInner, long cOuter, long cInner)
{
  int z = blockIdx.z;
  const half_t* Ab = A + (long)z * strideA;
  const half_t* Wb = W + (long)z * strideW;
  long coff = (long)(z / zInner) * cOuter + (long)(z % zInner) * cInner;

  int wave = threadIdx.x >> 5;
  int lane = threadIdx.x & 31;
  int mBase = blockIdx.y * 64 + (wave & 3) * 16;
  int nBase = blockIdx.x * 64 + (wave >> 2) * 32;
  int row = lane & 15;
  int hi  = (lane >> 4) & 1;

  const half_t* aPtr  = Ab + (long)(mBase + row) * lda + hi * 8;
  const half_t* bPtr0 = Wb + (long)(nBase + row) * ldw + hi * 8;
  const half_t* bPtr1 = bPtr0 + (long)16 * ldw;

  v8f acc0 = {}; v8f acc1 = {};
  for (int k = 0; k < K; k += 32) {
    v16h a  = load_frag_a(aPtr + k);
    v16h b0 = load_frag_b(bPtr0 + k);
    v16h b1 = load_frag_b(bPtr1 + k);
    acc0 = __builtin_amdgcn_wmma_f32_16x16x32_f16(false, a, false, b0, (short)0, acc0, false, false);
    acc1 = __builtin_amdgcn_wmma_f32_16x16x32_f16(false, a, false, b1, (short)0, acc1, false, false);
  }

  int cm = mBase + hi * 8;
  int cn = nBase + row;
  if (C_F16) {
    half_t* C = (half_t*)Cv + coff;
    #pragma unroll
    for (int i = 0; i < 8; ++i) {
      C[(long)(cm + i) * ldc + cn]      = (half_t)acc0[i];
      C[(long)(cm + i) * ldc + cn + 16] = (half_t)acc1[i];
    }
  } else {
    float* C = (float*)Cv + coff;
    float bv0 = BIAS ? bias[cn]      : 0.f;
    float bv1 = BIAS ? bias[cn + 16] : 0.f;
    #pragma unroll
    for (int i = 0; i < 8; ++i) {
      C[(long)(cm + i) * ldc + cn]      = acc0[i] + bv0;
      C[(long)(cm + i) * ldc + cn + 16] = acc1[i] + bv1;
    }
  }
}

// ---------------------------------------------------------------------------
// NT WMMA GEMM with TDM-staged W panel. The 64-row W panel shared by the whole
// block is DMA'd into double-buffered LDS by the Tensor Data Mover (wave 0
// issues, TENSORcnt-tracked, pipelined one K-block ahead). A streams from
// global; W fragments come from LDS (ds_load_b128). Requires K % TDM_KB == 0.
// ---------------------------------------------------------------------------
template<int C_F16, int BIAS>
__global__ __launch_bounds__(256) void gemm_nt_wmma_tdm(
    const half_t* __restrict__ A, const half_t* __restrict__ W,
    void* __restrict__ Cv, const float* __restrict__ bias,
    int K, int lda, int ldw, int ldc,
    long strideA, long strideW, int zInner, long cOuter, long cInner)
{
  __shared__ half_t Wlds[2][64 * TDM_KB];   // 2 x 16KB double buffer

  int z = blockIdx.z;
  const half_t* Ab = A + (long)z * strideA;
  const half_t* Wb = W + (long)z * strideW;
  long coff = (long)(z / zInner) * cOuter + (long)(z % zInner) * cInner;

  int wave = threadIdx.x >> 5;
  int lane = threadIdx.x & 31;
  int mBase = blockIdx.y * 64 + (wave & 3) * 16;
  int nLoc  = (wave >> 2) * 32;
  int nBase = blockIdx.x * 64 + nLoc;
  int row = lane & 15;
  int hi  = (lane >> 4) & 1;

  const half_t* aPtr = Ab + (long)(mBase + row) * lda + hi * 8;
  const half_t* bL0  = &Wlds[0][0] + (nLoc + row) * TDM_KB + hi * 8;
  const half_t* bL1  = bL0 + 16 * TDM_KB;
  const half_t* wPanel = Wb + (long)(blockIdx.x * 64) * ldw;
  unsigned ldsBase = (unsigned)(size_t)(&Wlds[0][0]);   // LDS aperture: low 32 bits = byte offset
  const unsigned bufBytes = 64u * TDM_KB * 2u;

  int nkb = K / TDM_KB;
  bool issuer = (threadIdx.x < 32);

  if (issuer) tdm_issue_panel(wPanel, ldw, ldsBase);

  v8f acc0 = {}; v8f acc1 = {};
  for (int kb = 0; kb < nkb; ++kb) {
    if (issuer) {
      if (kb + 1 < nkb) {
        tdm_issue_panel(wPanel + (long)(kb + 1) * TDM_KB,
                        ldw, ldsBase + (unsigned)((kb + 1) & 1) * bufBytes);
        __builtin_amdgcn_s_wait_tensorcnt(1);  // in-order TDM: kb's panel has landed
      } else {
        __builtin_amdgcn_s_wait_tensorcnt(0);
      }
    }
    __syncthreads();                           // panel kb visible to all waves
    const half_t* bp0 = bL0 + (kb & 1) * (64 * TDM_KB);
    const half_t* bp1 = bL1 + (kb & 1) * (64 * TDM_KB);
    const half_t* ap  = aPtr + (long)kb * TDM_KB;
    #pragma unroll
    for (int k = 0; k < TDM_KB; k += 32) {
      v16h a  = load_frag_a(ap + k);
      v16h b0 = load_frag_b(bp0 + k);
      v16h b1 = load_frag_b(bp1 + k);
      acc0 = __builtin_amdgcn_wmma_f32_16x16x32_f16(false, a, false, b0, (short)0, acc0, false, false);
      acc1 = __builtin_amdgcn_wmma_f32_16x16x32_f16(false, a, false, b1, (short)0, acc1, false, false);
    }
    __syncthreads();                           // all reads done before buffer is re-targeted
  }

  int cm = mBase + hi * 8;
  int cn = nBase + row;
  if (C_F16) {
    half_t* C = (half_t*)Cv + coff;
    #pragma unroll
    for (int i = 0; i < 8; ++i) {
      C[(long)(cm + i) * ldc + cn]      = (half_t)acc0[i];
      C[(long)(cm + i) * ldc + cn + 16] = (half_t)acc1[i];
    }
  } else {
    float* C = (float*)Cv + coff;
    float bv0 = BIAS ? bias[cn]      : 0.f;
    float bv1 = BIAS ? bias[cn + 16] : 0.f;
    #pragma unroll
    for (int i = 0; i < 8; ++i) {
      C[(long)(cm + i) * ldc + cn]      = acc0[i] + bv0;
      C[(long)(cm + i) * ldc + cn + 16] = acc1[i] + bv1;
    }
  }
}

// ---------------------------------------------------------------------------
// kqv (B,L,3*D) f16 -> k,q (B,H,L,hd) masked (q pre-scaled) and vT (B,H,hd,L)
// ---------------------------------------------------------------------------
__global__ void split_kqv_kernel(const half_t* __restrict__ kqv, const float* __restrict__ mask,
                                 half_t* __restrict__ q, half_t* __restrict__ k,
                                 half_t* __restrict__ vT,
                                 int B, int H, int L, int hd, float qscale)
{
  long idx = (long)blockIdx.x * blockDim.x + threadIdx.x;
  long total = (long)B * H * L * hd;
  if (idx >= total) return;
  int d = (int)(idx % hd);  long t = idx / hd;
  int l = (int)(t % L);     t /= L;
  int h = (int)(t % H);
  int b = (int)(t / H);
  int D = H * hd;
  float mv = mask[(long)b * L + l];
  long src = ((long)b * L + l) * (3 * D) + h * hd + d;
  float kv = (float)kqv[src]         * mv;
  float qv = (float)kqv[src + D]     * mv * qscale;
  float vv = (float)kqv[src + 2 * D] * mv;
  k[idx] = (half_t)kv;
  q[idx] = (half_t)qv;
  vT[(((long)b * H + h) * hd + d) * L + l] = (half_t)vv;
}

// ---------------------------------------------------------------------------
// landmark means: q_lm/k_lm[b,h,m,d] = sum_s q/k[b,h,m*seg+s,d] / denom[b,m]
// ---------------------------------------------------------------------------
__global__ void landmarks_kernel(const half_t* __restrict__ q, const half_t* __restrict__ k,
                                 const float* __restrict__ denom,
                                 half_t* __restrict__ qlm, half_t* __restrict__ klm,
                                 int H, int L, int nl, int seg, int hd)
{
  int blk = blockIdx.x;
  int d = threadIdx.x;
  int m = blk % nl;
  int h = (blk / nl) % H;
  int b = blk / (nl * H);
  long base = (((long)b * H + h) * L + (long)m * seg) * hd + d;
  float sq = 0.f, sk = 0.f;
  for (int s = 0; s < seg; ++s) {
    sq += (float)q[base + (long)s * hd];
    sk += (float)k[base + (long)s * hd];
  }
  float inv = 1.f / denom[b * nl + m];
  long o = (((long)b * H + h) * nl + m) * hd + d;
  qlm[o] = (half_t)(sq * inv);
  klm[o] = (half_t)(sk * inv);
}

// ---------------------------------------------------------------------------
// softmax over 64 columns, one wave per row, additive column mask.
// (row-mask terms are uniform per row -> softmax-invariant -> dropped)
// ---------------------------------------------------------------------------
__global__ __launch_bounds__(256) void softmax64_kernel(const float* __restrict__ in,
                                                        half_t* __restrict__ out,
                                                        const float* __restrict__ colmask,
                                                        long nrows, int rowsPerB)
{
  int wave = threadIdx.x >> 5, lane = threadIdx.x & 31;
  long row = (long)blockIdx.x * 8 + wave;
  if (row >= nrows) return;
  int b = (int)(row / rowsPerB);
  const float* rp = in + row * 64;
  const float* cm = colmask + (long)b * 64;
  float v0 = rp[lane]      + (1.f - cm[lane])      * NEGF;
  float v1 = rp[lane + 32] + (1.f - cm[lane + 32]) * NEGF;
  float mx = fmaxf(v0, v1);
  for (int o = 16; o > 0; o >>= 1) mx = fmaxf(mx, __shfl_xor(mx, o, 32));
  float e0 = __expf(v0 - mx), e1 = __expf(v1 - mx);
  float sm = e0 + e1;
  for (int o = 16; o > 0; o >>= 1) sm += __shfl_xor(sm, o, 32);
  float inv = 1.f / sm;
  out[row * 64 + lane]      = (half_t)(e0 * inv);
  out[row * 64 + lane + 32] = (half_t)(e1 * inv);
}

// ---------------------------------------------------------------------------
// softmax over wide rows (ncol), one 256-thread block per row.
// ---------------------------------------------------------------------------
__global__ __launch_bounds__(256) void softmax_wide_kernel(const float* __restrict__ in,
                                                           half_t* __restrict__ out,
                                                           const float* __restrict__ colmask,
                                                           int ncol, int rowsPerB)
{
  __shared__ float red[8];
  int row = blockIdx.x;
  int b = row / rowsPerB;
  const float* rp = in + (long)row * ncol;
  const float* cm = colmask + (long)b * ncol;
  half_t* op = out + (long)row * ncol;
  int tid = threadIdx.x, lane = tid & 31, wv = tid >> 5;

  float mx = -3.4e38f;
  for (int c = tid; c < ncol; c += 256) mx = fmaxf(mx, rp[c] + (1.f - cm[c]) * NEGF);
  for (int o = 16; o > 0; o >>= 1) mx = fmaxf(mx, __shfl_xor(mx, o, 32));
  if (lane == 0) red[wv] = mx;
  __syncthreads();
  float m0 = red[0];
  for (int i = 1; i < 8; ++i) m0 = fmaxf(m0, red[i]);
  __syncthreads();

  float sm = 0.f;
  for (int c = tid; c < ncol; c += 256) {
    float e = __expf(rp[c] + (1.f - cm[c]) * NEGF - m0);
    op[c] = (half_t)e;
    sm += e;
  }
  for (int o = 16; o > 0; o >>= 1) sm += __shfl_xor(sm, o, 32);
  if (lane == 0) red[wv] = sm;
  __syncthreads();
  float tot = 0.f;
  for (int i = 0; i < 8; ++i) tot += red[i];
  float inv = 1.f / tot;
  for (int c = tid; c < ncol; c += 256) op[c] = (half_t)((float)op[c] * inv);
}

// ---------------------------------------------------------------------------
// Moore-Penrose pinv of scores_2 per (b,h). One block per head, fp32 in LDS.
//   A = X@Z; C1 = 7A - A@A; C2 = 15A - A@C1; Z = 0.25*(13Z - Z@C2)
// ---------------------------------------------------------------------------
__global__ __launch_bounds__(256) void pinv_kernel(const half_t* __restrict__ qlm,
                                                   const half_t* __restrict__ klm,
                                                   const float* __restrict__ masklm,
                                                   float* __restrict__ zstar, int H)
{
  __shared__ float S[4][64 * 64];   // X, Z, T1, T2 (T1/T2 double as QL/KL loaders)
  float* X  = S[0];
  float* Z  = S[1];
  float* T1 = S[2];
  float* T2 = S[3];
  int bh = blockIdx.x;
  int b = bh / H;
  int tid = threadIdx.x;
  int r  = tid >> 2;
  int c0 = (tid & 3) * 16;
  float reg[16];

  for (int i = tid; i < 4096; i += 256) {
    T1[i] = (float)qlm[(long)bh * 4096 + i];
    T2[i] = (float)klm[(long)bh * 4096 + i];
  }
  __syncthreads();

  for (int j = 0; j < 16; ++j) {
    int c = c0 + j;
    float s = 0.f;
    for (int kk = 0; kk < 64; ++kk) s = fmaf(T1[r * 64 + kk], T2[c * 64 + kk], s);
    reg[j] = s + (1.f - masklm[b * 64 + c]) * NEGF;
  }
  __syncthreads();
  for (int j = 0; j < 16; ++j) X[r * 64 + c0 + j] = reg[j];
  __syncthreads();

  if (tid < 64) {
    float mx = -3.4e38f;
    for (int c = 0; c < 64; ++c) mx = fmaxf(mx, X[tid * 64 + c]);
    float sm = 0.f;
    for (int c = 0; c < 64; ++c) { float e = __expf(X[tid * 64 + c] - mx); X[tid * 64 + c] = e; sm += e; }
    float inv = 1.f / sm;
    for (int c = 0; c < 64; ++c) X[tid * 64 + c] *= inv;
  }
  __syncthreads();

  if (tid < 64) {
    float rs = 0.f, cs = 0.f;
    for (int c = 0; c < 64; ++c) { rs += fabsf(X[tid * 64 + c]); cs += fabsf(X[c * 64 + tid]); }
    T1[tid] = rs; T1[64 + tid] = cs;
  }
  __syncthreads();
  if (tid == 0) {
    float m1 = 0.f, m2 = 0.f;
    for (int i = 0; i < 64; ++i) { m1 = fmaxf(m1, T1[i]); m2 = fmaxf(m2, T1[64 + i]); }
    T1[128] = 1.f / (m1 * m2);
  }
  __syncthreads();
  float sc = T1[128];
  __syncthreads();

  for (int e = tid; e < 4096; e += 256) { int rr = e >> 6, cc = e & 63; Z[e] = X[cc * 64 + rr] * sc; }
  __syncthreads();

  for (int it = 0; it < 6; ++it) {
    for (int j = 0; j < 16; ++j) {
      int c = c0 + j; float s = 0.f;
      for (int kk = 0; kk < 64; ++kk) s = fmaf(X[r * 64 + kk], Z[kk * 64 + c], s);
      reg[j] = s;
    }
    __syncthreads();
    for (int j = 0; j < 16; ++j) T1[r * 64 + c0 + j] = reg[j];
    __syncthreads();
    for (int j = 0; j < 16; ++j) {
      int c = c0 + j; float s = 0.f;
      for (int kk = 0; kk < 64; ++kk) s = fmaf(T1[r * 64 + kk], T1[kk * 64 + c], s);
      reg[j] = 7.f * T1[r * 64 + c] - s;
    }
    __syncthreads();
    for (int j = 0; j < 16; ++j) T2[r * 64 + c0 + j] = reg[j];
    __syncthreads();
    for (int j = 0; j < 16; ++j) {
      int c = c0 + j; float s = 0.f;
      for (int kk = 0; kk < 64; ++kk) s = fmaf(T1[r * 64 + kk], T2[kk * 64 + c], s);
      reg[j] = 15.f * T1[r * 64 + c] - s;
    }
    __syncthreads();
    for (int j = 0; j < 16; ++j) T2[r * 64 + c0 + j] = reg[j];
    __syncthreads();
    for (int j = 0; j < 16; ++j) {
      int c = c0 + j; float s = 0.f;
      for (int kk = 0; kk < 64; ++kk) s = fmaf(Z[r * 64 + kk], T2[kk * 64 + c], s);
      reg[j] = 0.25f * (13.f * Z[r * 64 + c] - s);
    }
    __syncthreads();
    for (int j = 0; j < 16; ++j) Z[r * 64 + c0 + j] = reg[j];
    __syncthreads();
  }

  for (int e = tid; e < 4096; e += 256) zstar[(long)bh * 4096 + e] = Z[e];
}

// ---------------------------------------------------------------------------
// t2T[b,h,n,k] = (zstar @ t1)[k,n]  (transposed f16 store so s1@t2 stays NT)
// ---------------------------------------------------------------------------
__global__ __launch_bounds__(256) void zt1_kernel(const float* __restrict__ zstar,
                                                  const float* __restrict__ t1,
                                                  half_t* __restrict__ t2T)
{
  __shared__ float Zs[4096], Ts[4096];
  int bh = blockIdx.x;
  int tid = threadIdx.x;
  for (int i = tid; i < 4096; i += 256) {
    Zs[i] = zstar[(long)bh * 4096 + i];
    Ts[i] = t1[(long)bh * 4096 + i];
  }
  __syncthreads();
  int r = tid >> 2, c0 = (tid & 3) * 16;
  for (int j = 0; j < 16; ++j) {
    int c = c0 + j; float s = 0.f;
    for (int kk = 0; kk < 64; ++kk) s = fmaf(Zs[r * 64 + kk], Ts[kk * 64 + c], s);
    t2T[(long)bh * 4096 + (long)c * 64 + r] = (half_t)s;
  }
}

// ---------------------------------------------------------------------------
extern "C" void kernel_launch(void* const* d_in, const int* in_sizes, int n_in,
                              void* d_out, int out_size, void* d_ws, size_t ws_size,
                              hipStream_t stream)
{
  const int D = 1024, H = 16, hd = 64, nl = 64;
  const int L = 4096;                       // reference: (4, 4096, 1024)
  const int B = in_sizes[1] / L;            // attention_mask is (B, L)
  const long BL = (long)B * L;
  const int  BH = B * H;

  const float* x    = (const float*)d_in[0];
  const float* mask = (const float*)d_in[1];
  const float* wkqv = (const float*)d_in[2];
  const float* wout = (const float*)d_in[3];
  const float* bout = (const float*)d_in[4];
  float* outp = (float*)d_out;
  (void)n_in; (void)out_size; (void)ws_size;

  char* ws = (char*)d_ws;
  size_t off = 0;
  auto take = [&](size_t bytes) { size_t r = off; off += (bytes + 255) & ~(size_t)255; return r; };

  half_t* x16    = (half_t*)(ws + take(BL * D * sizeof(half_t)));        // reused as s1f16
  half_t* wkqv16 = (half_t*)(ws + take((size_t)3 * D * D * sizeof(half_t)));
  half_t* wout16 = (half_t*)(ws + take((size_t)D * D * sizeof(half_t)));
  size_t  kqvOff = take(BL * 3 * D * sizeof(half_t));
  half_t* kqv16  = (half_t*)(ws + kqvOff);                               // reused: outattn + s3f16
  half_t* q16    = (half_t*)(ws + take(BL * D * sizeof(half_t)));
  half_t* k16    = (half_t*)(ws + take(BL * D * sizeof(half_t)));
  half_t* vT16   = (half_t*)(ws + take(BL * D * sizeof(half_t)));
  half_t* qlm16  = (half_t*)(ws + take((size_t)BH * nl * hd * sizeof(half_t)));
  half_t* klm16  = (half_t*)(ws + take((size_t)BH * nl * hd * sizeof(half_t)));
  float*  denom  = (float*)(ws + take((size_t)B * nl * sizeof(float)));
  float*  masklm = (float*)(ws + take((size_t)B * nl * sizeof(float)));
  float*  sraw   = (float*)(ws + take((size_t)BH * L * nl * sizeof(float))); // s1raw then s3raw
  float*  t1     = (float*)(ws + take((size_t)BH * nl * hd * sizeof(float)));
  float*  zstar  = (float*)(ws + take((size_t)BH * nl * nl * sizeof(float)));
  half_t* t2T    = (half_t*)(ws + take((size_t)BH * nl * hd * sizeof(half_t)));
  half_t* s1f16   = x16;                                                  // x dead after kqv GEMM
  half_t* outattn = kqv16;                                                // kqv dead after split
  half_t* s3f16   = (half_t*)(ws + kqvOff + (size_t)BL * D * sizeof(half_t));

  dim3 blk(256);

  // 1. fp32 -> fp16 operand conversions
  cvt_kernel<<<8192, 256, 0, stream>>>(x, x16, BL * D);
  cvt_kernel<<<4096, 256, 0, stream>>>(wkqv, wkqv16, (long)3 * D * D);
  cvt_kernel<<<2048, 256, 0, stream>>>(wout, wout16, (long)D * D);

  // 2. landmark mask statistics
  mask_stats_kernel<<<(B * nl + 255) / 256, 256, 0, stream>>>(mask, denom, masklm, B, L, nl);

  // 3. kqv = x @ w_kqv^T  (16384 x 3072 x 1024) [TDM-staged W panel]
  gemm_nt_wmma_tdm<1, 0><<<dim3(3 * D / 64, BL / 64, 1), blk, 0, stream>>>(
      x16, wkqv16, kqv16, nullptr, D, D, D, 3 * D, 0L, 0L, 1, 0L, 0L);

  // 4. split into k,q (masked, q*1/sqrt(hd)) and vT
  split_kqv_kernel<<<(unsigned)((BL * D + 255) / 256), 256, 0, stream>>>(
      kqv16, mask, q16, k16, vT16, B, H, L, hd, 0.125f);

  // 5. landmark means
  landmarks_kernel<<<BH * nl, 64, 0, stream>>>(q16, k16, denom, qlm16, klm16, H, L, nl, L / nl, hd);

  // 6. s1raw = q @ k_lm^T   (per bh: 4096 x 64 x 64)
  gemm_nt_wmma<0, 0><<<dim3(nl / 64, L / 64, BH), blk, 0, stream>>>(
      q16, klm16, sraw, nullptr, hd, hd, hd, nl,
      (long)L * hd, (long)nl * hd, 1, (long)L * nl, 0L);

  // 7. scores_1 = softmax(s1raw) -> f16
  softmax64_kernel<<<(unsigned)((long)BH * L / 8), 256, 0, stream>>>(
      sraw, s1f16, masklm, (long)BH * L, H * L);

  // 8. s3raw = q_lm @ k^T   (per bh: 64 x 4096 x 64)
  gemm_nt_wmma<0, 0><<<dim3(L / 64, nl / 64, BH), blk, 0, stream>>>(
      qlm16, k16, sraw, nullptr, hd, hd, hd, L,
      (long)nl * hd, (long)L * hd, 1, (long)nl * L, 0L);

  // 9. scores_3 = softmax(s3raw) -> f16
  softmax_wide_kernel<<<BH * nl, 256, 0, stream>>>(sraw, s3f16, mask, L, H * nl);

  // 10. t1 = scores_3 @ v   (per bh: 64 x 64 x 4096, W = v^T) [TDM-staged]
  gemm_nt_wmma_tdm<0, 0><<<dim3(hd / 64, nl / 64, BH), blk, 0, stream>>>(
      s3f16, vT16, t1, nullptr, L, L, L, hd,
      (long)nl * L, (long)hd * L, 1, (long)nl * hd, 0L);

  // 11. z_star = pinv(softmax(q_lm @ k_lm^T))
  pinv_kernel<<<BH, 256, 0, stream>>>(qlm16, klm16, masklm, zstar, H);

  // 12. t2T = (z_star @ t1)^T -> f16
  zt1_kernel<<<BH, 256, 0, stream>>>(zstar, t1, t2T);

  // 13. outattn[b, l, h*hd + n] = (scores_1 @ t2)[l, n]   (per bh: 4096 x 64 x 64)
  gemm_nt_wmma<1, 0><<<dim3(hd / 64, L / 64, BH), blk, 0, stream>>>(
      s1f16, t2T, outattn, nullptr, nl, nl, nl, H * hd,
      (long)L * nl, (long)nl * hd, H, (long)L * H * hd, (long)hd);

  // 14. out = outattn @ w_out^T + b_out  (16384 x 1024 x 1024) [TDM-staged]
  gemm_nt_wmma_tdm<0, 1><<<dim3(D / 64, BL / 64, 1), blk, 0, stream>>>(
      outattn, wout16, outp, bout, D, D, D, D, 0L, 0L, 1, 0L, 0L);
}
